// AdaptiveSynapticDelayAttention_50405736186452
// MI455X (gfx1250) — matrix operand order repaired and checked by reading, not compiled, not run on hardware
//
#include <hip/hip_runtime.h>
#include <hip/hip_bf16.h>
#include <math.h>

// ---------------------------------------------------------------------------
// AdaptiveSynapticDelayAttention for MI455X (gfx1250, wave32, WMMA + TDM).
//
// Pipeline (f16 storage, f32 WMMA accumulate):
//   1. convert Wq/Wk/Wv/Wo -> f16
//   2. delay-gather spikes -> delayedH (f16)
//   3. Q = delayedH @ Wq^T, K = delayedH @ Wk^T      (WMMA NT GEMM, f16)
//      Vt = (delayedH @ Wv^T)^T                      (WMMA NT GEMM, f16 transposed store)
//   4. per batch b (scores/P buffers reused, L2-resident):
//        scores = Q_b K_b^T / sqrt(768)              (WMMA NT GEMM, f32)
//        P      = softmax_rows(scores)               (f16)
//        attn_b = P @ Vt_b^T                         (WMMA NT GEMM, f16)
//   5. out = attnH @ Wo^T + bo                       (WMMA NT GEMM, f32)
//
// All GEMM global->LDS tile traffic goes through the Tensor Data Mover
// (tensor_load_to_lds + s_wait_tensorcnt) with double-buffered LDS so the
// DMA of tile k+1 overlaps the WMMA math on tile k.
//
// mscores/membrane-potential path skipped: it adds a per-query constant along
// the softmax axis and softmax is shift-invariant => exact no-op.
// ---------------------------------------------------------------------------

typedef __attribute__((ext_vector_type(16))) _Float16 v16h;
typedef __attribute__((ext_vector_type(8)))  _Float16 v8h;
typedef __attribute__((ext_vector_type(8)))  float    v8f;
typedef __attribute__((ext_vector_type(4)))  unsigned int u32x4;
typedef __attribute__((ext_vector_type(8)))  int      i32x8;
typedef __attribute__((ext_vector_type(4)))  int      i32x4;

#define Bdim 8
#define Sdim 2048
#define Edim 768
#define NTOK (Bdim * Sdim)

#if defined(__has_builtin)
#if __has_builtin(__builtin_amdgcn_tensor_load_to_lds) && \
    __has_builtin(__builtin_amdgcn_s_wait_tensorcnt)
#define USE_TDM 1
#endif
#endif
#ifndef USE_TDM
#define USE_TDM 0
#endif

// ------------------------- small helper kernels ----------------------------

__global__ void delay_gather_kernel(const float* __restrict__ spikes,
                                    const int* __restrict__ delays,
                                    _Float16* __restrict__ out) {
    int e = blockIdx.x * blockDim.x + threadIdx.x;   // 0..767
    int t = blockIdx.y;                              // 0..2047
    int b = blockIdx.z;                              // 0..7
    int d = delays[e];                               // H==1 -> row 0
    int src = t - d;
    float v = (src >= 0) ? spikes[((size_t)b * Sdim + src) * Edim + e] : 0.0f;
    out[((size_t)b * Sdim + t) * Edim + e] = (_Float16)v;
}

__global__ void f32_to_f16_kernel(const float* __restrict__ in,
                                  _Float16* __restrict__ out, int n) {
    for (int i = blockIdx.x * blockDim.x + threadIdx.x; i < n;
         i += gridDim.x * blockDim.x)
        out[i] = (_Float16)in[i];
}

__global__ void softmax_rows_kernel(const float* __restrict__ S,
                                    _Float16* __restrict__ P, int ncols) {
    __shared__ float red[256];
    const int row = blockIdx.x;
    const int tid = threadIdx.x;
    const float* srow = S + (size_t)row * ncols;
    _Float16* prow = P + (size_t)row * ncols;

    float m = -3.4e38f;
    for (int c = tid; c < ncols; c += 256) m = fmaxf(m, srow[c]);
    red[tid] = m; __syncthreads();
    for (int s = 128; s > 0; s >>= 1) {
        if (tid < s) red[tid] = fmaxf(red[tid], red[tid + s]);
        __syncthreads();
    }
    const float rowmax = red[0];
    __syncthreads();

    float sum = 0.0f;
    for (int c = tid; c < ncols; c += 256) {
        float e = __expf(srow[c] - rowmax);
        sum += e;
        prow[c] = (_Float16)e;
    }
    red[tid] = sum; __syncthreads();
    for (int s = 128; s > 0; s >>= 1) {
        if (tid < s) red[tid] += red[tid + s];
        __syncthreads();
    }
    const float inv = 1.0f / red[0];
    __syncthreads();
    for (int c = tid; c < ncols; c += 256)
        prow[c] = (_Float16)((float)prow[c] * inv);
}

// --------------------------- TDM 2D tile load ------------------------------
// Builds the D# per CDNA5 ISA and issues
// tensor_load_to_lds: a [tileH x tileW] (halves) tile at gaddr with row
// stride strideHalves -> contiguous LDS at ldsOff. Wave-level op.

#if USE_TDM
__device__ inline void tdm_load_tile_2d(unsigned ldsOff, const _Float16* gaddr,
                                        unsigned tileW, unsigned tileH,
                                        unsigned long long strideHalves) {
    const unsigned long long ga = (unsigned long long)gaddr;
    u32x4 g0;
    g0[0] = 1u;                                    // count=1, user descriptor
    g0[1] = ldsOff;                                // lds_addr (bytes)
    g0[2] = (unsigned)ga;                          // global_addr[31:0]
    g0[3] = (unsigned)(ga >> 32) | 0x80000000u;    // global_addr[56:32], type=2
    const unsigned d0 = (unsigned)strideHalves;    // tensor_dim0 (OOB bound)
    const unsigned d1 = 1u << 28;                  // tensor_dim1 (OOB bound)
    i32x8 g1;
    g1[0] = (int)(1u << 16);                                    // data_size=2B
    g1[1] = (int)((d0 & 0xFFFFu) << 16);                        // dim0 lo16
    g1[2] = (int)((d0 >> 16) | ((d1 & 0xFFFFu) << 16));         // dim0 hi/dim1 lo
    g1[3] = (int)((d1 >> 16) | (tileW << 16));                  // dim1 hi, tile_dim0
    g1[4] = (int)(tileH & 0xFFFFu);                             // tile_dim1, tile_dim2=0
    g1[5] = (int)(unsigned)strideHalves;                        // dim0_stride lo32
    g1[6] = (int)((unsigned)(strideHalves >> 32) & 0xFFFFu);    // dim0_stride hi16
    g1[7] = 0;                                                  // dim1_stride hi
    const i32x4 z4 = {0, 0, 0, 0};
#if __clang_major__ >= 23
    const i32x8 z8 = {0, 0, 0, 0, 0, 0, 0, 0};
    __builtin_amdgcn_tensor_load_to_lds(g0, g1, z4, z4, z8, 0);
#else
    __builtin_amdgcn_tensor_load_to_lds(g0, g1, z4, z4, 0);
#endif
}
#endif

// ------------------------------ WMMA GEMM ----------------------------------
// C[M,N] = scale * A[M,K] x B[N,K]^T (+ bias[N])   -- always NT.
// OUTM: 0 = f16 row-major, 1 = f32 row-major, 2 = f16 transposed (C^T, ld=ldc)
// Tiles: 128x128x32 / workgroup, 256 threads = 8 wave32; wave w owns 64x32
// (wm=(w&1)*64 -> 4 M-tiles, wn=(w>>1)*32 -> 2 N-tiles, 8 accum fragments).
// LDS K-contiguous: As[buf][128][32], Bs[buf][128][32] (Bs row = N column),
// double-buffered; tiles staged by the TDM while WMMA consumes prior stage.
// Fragment layouts per CDNA5 ISA 7.12.2 (16-bit A 16x32, B 32x16, f32 C 16x16).

__device__ inline v16h frag_combine(v8h lo, v8h hi8) {
    v16h r;
#pragma unroll
    for (int i = 0; i < 8; ++i) { r[i] = lo[i]; r[8 + i] = hi8[i]; }
    return r;
}

template <int OUTM, bool ADD_BIAS>
__global__ __launch_bounds__(256) void gemm_wmma_kernel(
    const _Float16* __restrict__ A, int lda,
    const _Float16* __restrict__ B, int ldb,
    void* __restrict__ Cout, int ldc, const float* __restrict__ bias,
    int M, int N, int K, float scale) {
    constexpr int BM = 128, BN = 128, BK = 32;
    __shared__ _Float16 As[2][BM * BK];
    __shared__ _Float16 Bs[2][BN * BK];

    const int tid   = threadIdx.x;
    const int lane  = tid & 31;
    const int wave  = tid >> 5;
    const int lhalf = lane & 15;   // 0..15
    const int hi    = lane >> 4;   // 0 or 1
    const int wm    = (wave & 1) * 64;
    const int wn    = (wave >> 1) * 32;
    const int rowBase = blockIdx.y * BM;
    const int colBase = blockIdx.x * BN;
    const int nk = K / BK;

    const _Float16* Abase = A + (size_t)rowBase * lda;
    const _Float16* Bbase = B + (size_t)colBase * ldb;

    v8f acc[4][2] = {};

#if USE_TDM
    if (wave == 0) {   // stage 0 via TDM
        tdm_load_tile_2d((unsigned)(size_t)&As[0][0], Abase, BK, BM, (unsigned)lda);
        tdm_load_tile_2d((unsigned)(size_t)&Bs[0][0], Bbase, BK, BN, (unsigned)ldb);
    }
#endif

    for (int kt = 0; kt < nk; ++kt) {
        const int cur = kt & 1;
#if USE_TDM
        if (wave == 0) {
            if (kt + 1 < nk) {   // prefetch stage kt+1 into other buffer
                const int k1 = (kt + 1) * BK;
                tdm_load_tile_2d((unsigned)(size_t)&As[cur ^ 1][0],
                                 Abase + k1, BK, BM, (unsigned)lda);
                tdm_load_tile_2d((unsigned)(size_t)&Bs[cur ^ 1][0],
                                 Bbase + k1, BK, BN, (unsigned)ldb);
                // TDM ops complete in order: <=2 outstanding => stage kt landed
                __builtin_amdgcn_s_wait_tensorcnt((short)2);
            } else {
                __builtin_amdgcn_s_wait_tensorcnt((short)0);
            }
        }
        __syncthreads();
#else
        {   // fallback cooperative copy (16 halves / thread / tile)
            const int idx = tid * 16;
            const int r = idx >> 5, c = idx & 31;
            const int k0 = kt * BK;
            *(v16h*)&As[cur][idx] = *(const v16h*)(Abase + (size_t)r * lda + k0 + c);
            *(v16h*)&Bs[cur][idx] = *(const v16h*)(Bbase + (size_t)r * ldb + k0 + c);
        }
        __syncthreads();
#endif

        // ---- fragments from LDS ----
        v16h afrag[4];
#pragma unroll
        for (int mt = 0; mt < 4; ++mt) {
            const _Float16* p = &As[cur][(wm + mt * 16 + lhalf) * BK + hi * 8];
            afrag[mt] = frag_combine(*(const v8h*)p, *(const v8h*)(p + 16));
        }
        v16h bfrag[2];
#pragma unroll
        for (int nt = 0; nt < 2; ++nt)
            bfrag[nt] = frag_combine(*(const v8h*)&Bs[cur][(wn + nt * 16 + lhalf) * BK + hi * 8], *(const v8h*)&Bs[cur][(wn + nt * 16 + lhalf) * BK + hi * 8 + 16]);

        // ---- 8 WMMA per wave per K step ----
#pragma unroll
        for (int mt = 0; mt < 4; ++mt)
#pragma unroll
            for (int nt = 0; nt < 2; ++nt)
                acc[mt][nt] = __builtin_amdgcn_wmma_f32_16x16x32_f16(
                    false, afrag[mt], false, bfrag[nt],
                    (short)0, acc[mt][nt], false, false);
        __syncthreads();
    }

    // ---- epilogue: f32 C 16x16 layout: VGPR r, lane -> M = r + 8*hi, N = lhalf
    float* Cf = (float*)Cout;
    _Float16* Ch = (_Float16*)Cout;
#pragma unroll
    for (int mt = 0; mt < 4; ++mt)
#pragma unroll
        for (int nt = 0; nt < 2; ++nt) {
            const int row0 = rowBase + wm + mt * 16 + hi * 8;
            const int col  = colBase + wn + nt * 16 + lhalf;
            if (OUTM == 2) {
                // transposed f16 store: 8 consecutive rows are contiguous
                v8h pk;
#pragma unroll
                for (int r = 0; r < 8; ++r)
                    pk[r] = (_Float16)(acc[mt][nt][r] * scale);
                *(v8h*)&Ch[(size_t)col * ldc + row0] = pk;
            } else {
#pragma unroll
                for (int r = 0; r < 8; ++r) {
                    float v = acc[mt][nt][r] * scale;
                    if (ADD_BIAS) v += bias[col];
                    const size_t off = (size_t)(row0 + r) * ldc + col;
                    if (OUTM == 0) Ch[off] = (_Float16)v;
                    else           Cf[off] = v;
                }
            }
        }
}

// ------------------------------- launcher ----------------------------------

extern "C" void kernel_launch(void* const* d_in, const int* in_sizes, int n_in,
                              void* d_out, int out_size, void* d_ws, size_t ws_size,
                              hipStream_t stream) {
    const float* spikes = (const float*)d_in[0];
    const int*   delays = (const int*)d_in[1];
    // d_in[2] membrane_potentials, d_in[3] decay_constant: provably no effect
    const float* Wq = (const float*)d_in[4];
    const float* Wk = (const float*)d_in[5];
    const float* Wv = (const float*)d_in[6];
    const float* Wo = (const float*)d_in[7];
    const float* bo = (const float*)d_in[8];
    float* out = (float*)d_out;

    // workspace carve-up
    char* ws = (char*)d_ws;
    const size_t tokBytes = (size_t)NTOK * Edim * sizeof(_Float16);   // 25.2 MB
    const size_t wBytes   = (size_t)Edim * Edim * sizeof(_Float16);   // 1.2 MB
    _Float16* delayedH = (_Float16*)ws;                 ws += tokBytes;
    _Float16* qH       = (_Float16*)ws;                 ws += tokBytes;
    _Float16* kH       = (_Float16*)ws;                 ws += tokBytes;
    _Float16* vT       = (_Float16*)ws;                 ws += tokBytes; // [768][16384]
    _Float16* attnH    = (_Float16*)ws;                 ws += tokBytes;
    _Float16* WqH      = (_Float16*)ws;                 ws += wBytes;
    _Float16* WkH      = (_Float16*)ws;                 ws += wBytes;
    _Float16* WvH      = (_Float16*)ws;                 ws += wBytes;
    _Float16* WoH      = (_Float16*)ws;                 ws += wBytes;
    float*    scoresF  = (float*)ws;    ws += (size_t)Sdim * Sdim * sizeof(float);
    _Float16* Pf       = (_Float16*)ws; ws += (size_t)Sdim * Sdim * sizeof(_Float16);

    const float invSqrtD = 0.03608439182435161f;   // 1/sqrt(768)

    // 1. weight conversion
    {
        dim3 g(64), blk(256);
        f32_to_f16_kernel<<<g, blk, 0, stream>>>(Wq, WqH, Edim * Edim);
        f32_to_f16_kernel<<<g, blk, 0, stream>>>(Wk, WkH, Edim * Edim);
        f32_to_f16_kernel<<<g, blk, 0, stream>>>(Wv, WvH, Edim * Edim);
        f32_to_f16_kernel<<<g, blk, 0, stream>>>(Wo, WoH, Edim * Edim);
    }
    // 2. delay gather
    {
        dim3 g(Edim / 256, Sdim, Bdim), blk(256);
        delay_gather_kernel<<<g, blk, 0, stream>>>(spikes, delays, delayedH);
    }
    // 3. projections: (16384x768) x (768x768)^T ; V stored transposed
    {
        dim3 g(Edim / 128, NTOK / 128), blk(256);
        gemm_wmma_kernel<0, false><<<g, blk, 0, stream>>>(
            delayedH, Edim, WqH, Edim, qH, Edim, nullptr, NTOK, Edim, Edim, 1.0f);
        gemm_wmma_kernel<0, false><<<g, blk, 0, stream>>>(
            delayedH, Edim, WkH, Edim, kH, Edim, nullptr, NTOK, Edim, Edim, 1.0f);
        gemm_wmma_kernel<2, false><<<g, blk, 0, stream>>>(
            delayedH, Edim, WvH, Edim, vT, NTOK, nullptr, NTOK, Edim, Edim, 1.0f);
    }
    // 4. attention per batch
    for (int b = 0; b < Bdim; ++b) {
        const size_t boff = (size_t)b * Sdim * Edim;
        {   // scores = Q K^T / sqrt(d)
            dim3 g(Sdim / 128, Sdim / 128), blk(256);
            gemm_wmma_kernel<1, false><<<g, blk, 0, stream>>>(
                qH + boff, Edim, kH + boff, Edim, scoresF, Sdim, nullptr,
                Sdim, Sdim, Edim, invSqrtD);
        }
        {   // P = softmax(scores)  (mscores omitted: softmax shift-invariant)
            softmax_rows_kernel<<<dim3(Sdim), dim3(256), 0, stream>>>(
                scoresF, Pf, Sdim);
        }
        {   // attn = P V = P x (Vt_b)^T ; Vt_b rows stride NTOK, col offset b*Sdim
            dim3 g(Edim / 128, Sdim / 128), blk(256);
            gemm_wmma_kernel<0, false><<<g, blk, 0, stream>>>(
                Pf, Sdim, vT + (size_t)b * Sdim, NTOK, attnH + boff, Edim,
                nullptr, Sdim, Edim, Sdim, 1.0f);
        }
    }
    // 5. output projection + bias -> f32 out
    {
        dim3 g(Edim / 128, NTOK / 128), blk(256);
        gemm_wmma_kernel<1, true><<<g, blk, 0, stream>>>(
            attnH, Edim, WoH, Edim, out, Edim, bo, NTOK, Edim, Edim, 1.0f);
    }
    (void)in_sizes; (void)n_in; (void)out_size; (void)ws_size;
}
